// Head_28845000360156
// MI455X (gfx1250) — matrix operand order repaired and checked by reading, not compiled, not run on hardware
//
#include <hip/hip_runtime.h>
#include <hip/hip_bf16.h>

// MI455X / gfx1250, wave32, WMMA f16 16x16x32 path.
//
// Reference: single attention head with softmax over the QUERY axis (dim=1):
//   q,k,v = x@Wq, x@Wk, x@Wv ; S = q k^T * 384^-0.5 ; causal mask (s<=t valid)
//   P[:,s] = softmax_t(S[:,s])  (normalized per key-column over t in [s,255])
//   out = P @ v
//
// One workgroup per batch element b (512 blocks x 256 threads = 8 waves).

typedef _Float16 v16h __attribute__((ext_vector_type(16)));
typedef float    v8f  __attribute__((ext_vector_type(8)));

#define NB 512
#define NT 256
#define NC 384
#define HS 16

#define P_PITCH 258   // halfs/row of P: 129 dwords (odd) -> conflict-free column access
#define QP      24    // halfs/row of Q/K/V tiles (48B rows, 16B-aligned frag loads)

#define SP_BYTES   (NT * P_PITCH * 2)   // 132096
#define SQKV_BYTES (NT * QP * 2)        // 12288 each
#define NKSTEP     (NC / 32)            // 12 K-steps for the projections
#define WFRAG_N    (NKSTEP * 3 * 32 * 16)  // pre-swizzled W fragments (halfs) = 18432

static __device__ __forceinline__ v8f wmma_f16(v16h a, v16h b, v8f c) {
  // v_wmma_f32_16x16x32_f16  (neg_a, A, neg_b, B, c_mod, C, reuse_a, reuse_b)
  return __builtin_amdgcn_wmma_f32_16x16x32_f16(false, a, false, b, (short)0, c, false, false);
}

union F16x16 { uint4 u[2]; v16h h; };
union F16x8  { uint4 u;    _Float16 h[8]; };

__global__ __launch_bounds__(256, 1)
void head_attn_kernel(const float* __restrict__ x,
                      const float* __restrict__ wk,
                      const float* __restrict__ wq,
                      const float* __restrict__ wv,
                      float* __restrict__ out) {
  extern __shared__ char smem[];
  _Float16* sP = (_Float16*)smem;                 // [NT][P_PITCH] f16: S, then exp(S-m)
  _Float16* sQ = (_Float16*)(smem + SP_BYTES);    // [NT][QP] f16 (pre-scaled by 384^-0.5)
  _Float16* sK = sQ + NT * QP;
  _Float16* sV = sK + NT * QP;                    // later rescaled by 1/denom[s]
  _Float16* sWf = sP;                             // phase-1 only: swizzled W frags (36KB),
                                                  // aliases sP (not live until phase 2)
  const int tid  = threadIdx.x;
  const int lane = tid & 31;
  const int wave = tid >> 5;
  const int b    = blockIdx.x;

  const int m   = lane & 15;          // A-frag: row M; B/C-frag: col N  (per ISA layouts)
  const int hi  = lane >> 4;          // lane half select
  const int kbA = hi ? 8  : 0;        // 16-bit A layout: K base for this lane half
  const int kbB = hi ? 16 : 0;        // contiguous-order K base (unused; B frags use the kbA order)

  // ---- Phase 0: pre-swizzle Wq/Wk/Wv into per-lane B-fragment layout (shared by all waves).
  // frag element j = ((step*3+mat)*32 + l)*16 + i  ->  W[step*32 + 8*(l>>4) + (i&7) + 16*(i>>3)][l&15]
  for (int j = tid; j < WFRAG_N; j += 256) {
    const int i   = j & 15;
    const int l   = (j >> 4) & 31;
    const int mat = (j >> 9) % 3;
    const int st  = (j >> 9) / 3;
    const int c   = st * 32 + ((l >> 4) ? 8 : 0) + (i & 7) + ((i >> 3) << 4);
    const float* wsel = (mat == 0) ? wq : (mat == 1) ? wk : wv;
    sWf[j] = (_Float16)wsel[c * HS + (l & 15)];
  }
  __syncthreads();

  // ================= Phase 1: projections via WMMA (K=384 in 12 steps of 32) ============
  const float scale = 0.051031036307982884f;  // 384^-0.5 (reference scales by n_embed)
  v8f accQ0 = {}, accK0 = {}, accV0 = {};
  v8f accQ1 = {}, accK1 = {}, accV1 = {};
  const int t0a = wave * 32;          // this wave's two 16-row output tiles
  const int t0b = t0a + 16;
  const size_t xrow0 = ((size_t)b * NT + t0a + m) * NC;
  const size_t xrow1 = ((size_t)b * NT + t0b + m) * NC;

  for (int step = 0; step < NKSTEP; ++step) {
    const int c0 = step * 32;
    if (c0 + 32 < NC) {               // gfx1250 global_prefetch_b8 of next x slab
      __builtin_prefetch(x + xrow0 + c0 + 32, 0, 1);
      __builtin_prefetch(x + xrow1 + c0 + 32, 0, 1);
    }
    // B fragments: two ds_load_b128 each from the pre-swizzled pool
    F16x16 Fq, Fk, Fv;
    {
      const uint4* pq = (const uint4*)(sWf + (((step * 3 + 0) * 32 + lane) << 4));
      const uint4* pk = (const uint4*)(sWf + (((step * 3 + 1) * 32 + lane) << 4));
      const uint4* pv = (const uint4*)(sWf + (((step * 3 + 2) * 32 + lane) << 4));
      Fq.u[0] = pq[0]; Fq.u[1] = pq[1];
      Fk.u[0] = pk[0]; Fk.u[1] = pk[1];
      Fv.u[0] = pv[0]; Fv.u[1] = pv[1];
    }
    // A fragments: lanes 0-15 K in {0..7,16..23}, lanes 16-31 K in {8..15,24..31}
    const float4* pa0 = (const float4*)(x + xrow0 + c0 + kbA);
    const float4* pa1 = (const float4*)(x + xrow0 + c0 + 16 + kbA);
    float4 f0 = pa0[0], f1 = pa0[1], f2 = pa1[0], f3 = pa1[1];
    v16h a0;
    a0[0]=(_Float16)f0.x;  a0[1]=(_Float16)f0.y;  a0[2]=(_Float16)f0.z;  a0[3]=(_Float16)f0.w;
    a0[4]=(_Float16)f1.x;  a0[5]=(_Float16)f1.y;  a0[6]=(_Float16)f1.z;  a0[7]=(_Float16)f1.w;
    a0[8]=(_Float16)f2.x;  a0[9]=(_Float16)f2.y;  a0[10]=(_Float16)f2.z; a0[11]=(_Float16)f2.w;
    a0[12]=(_Float16)f3.x; a0[13]=(_Float16)f3.y; a0[14]=(_Float16)f3.z; a0[15]=(_Float16)f3.w;
    const float4* pb0 = (const float4*)(x + xrow1 + c0 + kbA);
    const float4* pb1 = (const float4*)(x + xrow1 + c0 + 16 + kbA);
    float4 g0 = pb0[0], g1 = pb0[1], g2 = pb1[0], g3 = pb1[1];
    v16h a1;
    a1[0]=(_Float16)g0.x;  a1[1]=(_Float16)g0.y;  a1[2]=(_Float16)g0.z;  a1[3]=(_Float16)g0.w;
    a1[4]=(_Float16)g1.x;  a1[5]=(_Float16)g1.y;  a1[6]=(_Float16)g1.z;  a1[7]=(_Float16)g1.w;
    a1[8]=(_Float16)g2.x;  a1[9]=(_Float16)g2.y;  a1[10]=(_Float16)g2.z; a1[11]=(_Float16)g2.w;
    a1[12]=(_Float16)g3.x; a1[13]=(_Float16)g3.y; a1[14]=(_Float16)g3.z; a1[15]=(_Float16)g3.w;

    accQ0 = wmma_f16(a0, Fq.h, accQ0);  accQ1 = wmma_f16(a1, Fq.h, accQ1);
    accK0 = wmma_f16(a0, Fk.h, accK0);  accK1 = wmma_f16(a1, Fk.h, accK1);
    accV0 = wmma_f16(a0, Fv.h, accV0);  accV1 = wmma_f16(a1, Fv.h, accV1);
  }
  __syncthreads();   // sWf (aliasing sP) dead after this point

  // C/D layout: VGPR r -> row r + 8*hi, col = m
#pragma unroll
  for (int r = 0; r < 8; ++r) {
    const int r0 = t0a + r + hi * 8;
    const int r1 = t0b + r + hi * 8;
    sQ[r0 * QP + m] = (_Float16)(accQ0[r] * scale);
    sK[r0 * QP + m] = (_Float16)accK0[r];
    sV[r0 * QP + m] = (_Float16)accV0[r];
    sQ[r1 * QP + m] = (_Float16)(accQ1[r] * scale);
    sK[r1 * QP + m] = (_Float16)accK1[r];
    sV[r1 * QP + m] = (_Float16)accV1[r];
  }

  // ---- zero P so causal/untouched tiles read as p==0 in the O-pass WMMA
  {
    uint32_t* p32 = (uint32_t*)sP;
    for (int i = tid; i < SP_BYTES / 4; i += 256) p32[i] = 0u;
  }
  __syncthreads();

  // ================= Phase 2: S = (q*scale) k^T on the 136 causal tiles ================
  // d-contraction is 16, zero-padded to the x32 shape per the ISA A/B layouts.
  for (int idx = wave; idx < 136; idx += 8) {       // 17 tiles per wave, wave-uniform
    int ti = 0;
    while ((ti + 1) * (ti + 2) / 2 <= idx) ++ti;
    const int si = idx - ti * (ti + 1) / 2;

    // A: one uint4 LDS read (lane half selects d 0..7 vs 8..15); K>=16 stays zero
    F16x8 Aq;
    Aq.u = ((const uint4*)(sQ + (ti * 16 + m) * QP))[hi];
    v16h a = {};
#pragma unroll
    for (int i = 0; i < 8; ++i) a[i] = Aq.h[i];

    // B[d][s] = K[s][d]: two uint4 reads; lane half selects d 0..7 vs 8..15; K>=16 -> zero
    F16x16 Bk;
    {
      const uint4* kr = (const uint4*)(sK + (si * 16 + m) * QP);
      Bk.u[0] = kr[0]; Bk.u[1] = kr[1];
    }
    v16h bb;
#pragma unroll
    for (int i = 0; i < 16; ++i) bb[i] = (i < 8) ? Bk.h[i + hi * 8] : (_Float16)0;

    v8f c = {};
    v8f s = wmma_f16(a, bb, c);

    const int scol = si * 16 + m;
#pragma unroll
    for (int r = 0; r < 8; ++r) {
      const int t = ti * 16 + r + hi * 8;
      sP[t * P_PITCH + scol] = (scol <= t) ? (_Float16)s[r] : (_Float16)0.0f;
    }
  }
  __syncthreads();

  // ======== Phase 3: per-KEY-column softmax over t (dim=1), denom folded into V ========
  {
    const int s = tid;                              // thread owns key-column s
    float mx = -3.0e38f;
    for (int t = s; t < NT; ++t)
      mx = fmaxf(mx, (float)sP[t * P_PITCH + s]);
    float sum = 0.0f;
    for (int t = s; t < NT; ++t) {
      const float p = __expf((float)sP[t * P_PITCH + s] - mx);
      sum += p;
      sP[t * P_PITCH + s] = (_Float16)p;
    }
    const float inv = 1.0f / sum;
#pragma unroll
    for (int d = 0; d < HS; ++d)
      sV[s * QP + d] = (_Float16)((float)sV[s * QP + d] * inv);
  }
  __syncthreads();

  // ================= Phase 4: out = P @ V'  (contraction over s, K=32 steps) ===========
#pragma unroll
  for (int w2 = 0; w2 < 2; ++w2) {
    const int ti = wave * 2 + w2;
    v8f acc = {};
    const int nsteps = (ti + 2) >> 1;               // skip all-zero future-key blocks
    const uint* prow = (const uint*)(sP + (ti * 16 + m) * P_PITCH);  // 4B-aligned rows
    for (int ks = 0; ks < nsteps; ++ks) {
      const int c0 = ks * 32;
      // A frag: 8 x b32 LDS reads (rows are only 4B-aligned due to odd-dword pitch)
      union { uint u[4]; _Float16 h[8]; } A0, A1;
#pragma unroll
      for (int j = 0; j < 4; ++j) {
        A0.u[j] = prow[((c0 + kbA) >> 1) + j];
        A1.u[j] = prow[((c0 + 16 + kbA) >> 1) + j];
      }
      v16h a;
#pragma unroll
      for (int i = 0; i < 8; ++i) { a[i] = A0.h[i]; a[i + 8] = A1.h[i]; }
      v16h bb;
#pragma unroll
      for (int i = 0; i < 16; ++i)
        bb[i] = sV[(c0 + kbA + (i & 7) + ((i >> 3) << 4)) * QP + m];        // B[k][n] = V'[c0+k][n]
      acc = wmma_f16(a, bb, acc);
    }
#pragma unroll
    for (int r = 0; r < 8; ++r) {
      const int t = ti * 16 + r + hi * 8;
      out[((size_t)b * NT + t) * HS + m] = acc[r];  // lanes 0-15 write 64B runs
    }
  }
}

extern "C" void kernel_launch(void* const* d_in, const int* in_sizes, int n_in,
                              void* d_out, int out_size, void* d_ws, size_t ws_size,
                              hipStream_t stream) {
  const float* x  = (const float*)d_in[0];
  const float* wk = (const float*)d_in[1];   // k = x @ w_key
  const float* wq = (const float*)d_in[2];   // q = x @ w_query
  const float* wv = (const float*)d_in[3];   // v = x @ w_value
  float* out = (float*)d_out;                // [512,256,16] f32

  const size_t shmem = (size_t)SP_BYTES + 3 * (size_t)SQKV_BYTES;  // ~165 KB of 320 KB/WGP
  head_attn_kernel<<<NB, 256, shmem, stream>>>(x, wk, wq, wv, out);
}
